// AttentionHead_27462020891179
// MI455X (gfx1250) — matrix operand order repaired and checked by reading, not compiled, not run on hardware
//
#include <hip/hip_runtime.h>
#include <hip/hip_bf16.h>

#define Bc 4
#define Tc 4096
#define Cc 1024
#define Hc 64

typedef __attribute__((ext_vector_type(16))) _Float16 v16h;
typedef __attribute__((ext_vector_type(8)))  _Float16 v8h;
typedef __attribute__((ext_vector_type(8)))  float    v8f;
typedef __attribute__((ext_vector_type(4)))  float    v4f;

// Build a v16h A-operand from two contiguous 16B chunks (A-layout has the
// 16 f16 per lane split as K{k0..k0+7} and K{k0+16..k0+23}).
__device__ __forceinline__ v16h load2x8h(const _Float16* lo, const _Float16* hi) {
  v8h a = *(const v8h*)lo;
  v8h b = *(const v8h*)hi;
  return __builtin_shufflevector(a, b, 0,1,2,3,4,5,6,7,8,9,10,11,12,13,14,15);
}

__device__ __forceinline__ v8f wmma_f16(v16h a, v16h b, v8f c) {
  return __builtin_amdgcn_wmma_f32_16x16x32_f16(false, a, false, b, (short)0, c,
                                                false, false);
}

// ---------------------------------------------------------------------------
// Kernel 1: WT[w][h][c] = (f16) W_w[c][h]   (B-operand friendly layout)
// ---------------------------------------------------------------------------
__global__ void __launch_bounds__(256, 1)
wt_kernel(const float* __restrict__ Wq, const float* __restrict__ Wk,
          const float* __restrict__ Wv, _Float16* __restrict__ WT) {
  int i = blockIdx.x * blockDim.x + threadIdx.x;   // [w][h][c], 3*64*1024 total
  int c = i & (Cc - 1);
  int h = (i >> 10) & (Hc - 1);
  int w = i >> 16;
  const float* W = (w == 0) ? Wq : ((w == 1) ? Wk : Wv);
  WT[i] = (_Float16)W[c * Hc + h];
}

// ---------------------------------------------------------------------------
// Kernel 2: projection.  One wave computes 16 rows of q,k (f16 [B*T][64]) and
// vT (f16 [B][64][T]).  All 24 B loads + 4 A-prefetch loads are pinned before
// the WMMA chain with a sched_barrier so they issue as one clause into
// distinct registers.  launch_bounds(128,1) gives the allocator the full
// VGPR file (~230 live regs, no spills).
// ---------------------------------------------------------------------------
__global__ void __launch_bounds__(128, 1)
proj_kernel(const float* __restrict__ x, const _Float16* __restrict__ WT,
            _Float16* __restrict__ q, _Float16* __restrict__ k,
            _Float16* __restrict__ vT) {
  __shared__ _Float16 vtile[4][Hc][16];        // per-wave vT transpose staging
  const int lane = threadIdx.x & 31;
  const int wave = threadIdx.x >> 5;
  const int tile = blockIdx.x * 4 + wave;      // 0..1023
  const int row0 = tile * 16;
  const int lrow = lane & 15;
  const int hi   = lane >> 4;                  // 0 or 1
  const int row  = row0 + lrow;

  v8f acc[12];                                 // [w][hgroup]
  #pragma unroll
  for (int i = 0; i < 12; ++i) acc[i] = (v8f){};

  const float* xr = x + (size_t)row * Cc;
  // preload A operand for c0 = 0
  v4f a0 = *(const v4f*)(xr + hi * 8);
  v4f a1 = *(const v4f*)(xr + hi * 8 + 4);
  v4f a2 = *(const v4f*)(xr + hi * 8 + 16);
  v4f a3 = *(const v4f*)(xr + hi * 8 + 20);

  for (int c0 = 0; c0 < Cc; c0 += 32) {
    // issue all 12 B-tile loads into distinct registers (one clause)
    v16h Bt[12];
    #pragma unroll
    for (int w = 0; w < 3; ++w) {
      #pragma unroll
      for (int hg = 0; hg < 4; ++hg) {
        const _Float16* bp =
            WT + ((size_t)w * Hc + hg * 16 + lrow) * Cc + c0 + hi * 8;
        Bt[w * 4 + hg] = load2x8h(bp, bp + 16);
      }
    }
    // prefetch next iteration's A (clamped; redundant reload on last iter)
    const int cn = (c0 + 32 < Cc) ? (c0 + 32) : c0;
    const int ca = cn + hi * 8;
    v4f n0 = *(const v4f*)(xr + ca);
    v4f n1 = *(const v4f*)(xr + ca + 4);
    v4f n2 = *(const v4f*)(xr + ca + 16);
    v4f n3 = *(const v4f*)(xr + ca + 20);

    // nothing crosses: all loads above issue before any WMMA below
    __builtin_amdgcn_sched_barrier(0);

    // convert current A: elements 0-7 = K c0+8*hi+0..7 ; 8-15 = +16
    v16h A;
    #pragma unroll
    for (int i = 0; i < 4; ++i) {
      A[i]      = (_Float16)a0[i];
      A[4 + i]  = (_Float16)a1[i];
      A[8 + i]  = (_Float16)a2[i];
      A[12 + i] = (_Float16)a3[i];
    }
    #pragma unroll
    for (int j = 0; j < 12; ++j) acc[j] = wmma_f16(A, Bt[j], acc[j]);

    a0 = n0; a1 = n1; a2 = n2; a3 = n3;
  }

  // D-layout: lane = col n (h), element g = row g+8*hi.
  // qscale folds 1/sqrt(H) AND log2(e): softmax is computed in base 2.
  const float qscale = 0.125f * 1.44269504088896340736f;
  const int b  = row0 / Tc;
  const int t0 = row0 % Tc;
  #pragma unroll
  for (int hg = 0; hg < 4; ++hg) {
    const int h = hg * 16 + lrow;
    #pragma unroll
    for (int g = 0; g < 8; ++g) {
      const int r = row0 + g + 8 * hi;
      q[(size_t)r * Hc + h] = (_Float16)(acc[0 * 4 + hg][g] * qscale);
      k[(size_t)r * Hc + h] = (_Float16)(acc[1 * 4 + hg][g]);
      vtile[wave][h][g + 8 * hi] = (_Float16)acc[2 * 4 + hg][g];
    }
  }
  __syncthreads();                             // uniform path, all waves reach
  // coalesced vT stores: each lane writes 2 full h-rows (16 t = 32B each)
  #pragma unroll
  for (int rr = 0; rr < 2; ++rr) {
    const int h = lane + rr * 32;
    v8h lo  = *(const v8h*)&vtile[wave][h][0];
    v8h hi8 = *(const v8h*)&vtile[wave][h][8];
    _Float16* dst = vT + ((size_t)b * Hc + h) * Tc + t0;
    *(v8h*)dst       = lo;
    *(v8h*)(dst + 8) = hi8;
  }
}

// ---------------------------------------------------------------------------
// Kernel 3: causal flash attention.  One wave owns 32 queries (two 16-col Q^T
// tiles sharing every K/V load).  Per 32-key block: 8 wmma for S^T, base-2
// online softmax in-lane, P lands directly in A-layout, 8 wmma for P*V.
// Next block's K/V prefetch is pinned before the compute with sched_barrier;
// loop unrolled x2 so cur/nxt rotation is register renaming, not moves.
// launch_bounds(64,1): ~300 live VGPRs, no spills.
// ---------------------------------------------------------------------------
struct KVRegs { v16h ka00, ka01, ka10, ka11, vb0, vb1, vb2, vb3; };

__device__ __forceinline__ KVRegs load_kv(const _Float16* kb, const _Float16* vb,
                                          int s0, int lrow, int hi) {
  KVRegs r;
  const _Float16* k0r = kb + (size_t)(s0 + lrow) * Hc;
  const _Float16* k1r = kb + (size_t)(s0 + 16 + lrow) * Hc;
  const int ca = hi * 8;
  r.ka00 = load2x8h(k0r + ca,      k0r + ca + 16);
  r.ka01 = load2x8h(k0r + 32 + ca, k0r + 32 + ca + 16);
  r.ka10 = load2x8h(k1r + ca,      k1r + ca + 16);
  r.ka11 = load2x8h(k1r + 32 + ca, k1r + 32 + ca + 16);
  const _Float16* vp = vb + (size_t)lrow * Tc + s0 + hi * 8;
  r.vb0 = load2x8h(vp, vp + 16);
  r.vb1 = load2x8h(vp + (size_t)16 * Tc, vp + (size_t)16 * Tc + 16);
  r.vb2 = load2x8h(vp + (size_t)32 * Tc, vp + (size_t)32 * Tc + 16);
  r.vb3 = load2x8h(vp + (size_t)48 * Tc, vp + (size_t)48 * Tc + 16);
  return r;
}

// base-2 online softmax: S is already scaled by log2(e)/sqrt(H)
__device__ __forceinline__ void online_softmax(v8f& S0, v8f& S1, float& m, float& l,
                                               v16h& P, float& alpha) {
  float bm = S0[0];
  #pragma unroll
  for (int g = 0; g < 8; ++g) { bm = fmaxf(bm, S0[g]); bm = fmaxf(bm, S1[g]); }
  bm = fmaxf(bm, __shfl_xor(bm, 16, 32));
  const float mn = fmaxf(m, bm);
  alpha = __builtin_amdgcn_exp2f(m - mn);
  float ps = 0.0f;
  #pragma unroll
  for (int g = 0; g < 8; ++g) {
    float p0 = __builtin_amdgcn_exp2f(S0[g] - mn);
    float p1 = __builtin_amdgcn_exp2f(S1[g] - mn);
    ps += p0 + p1;
    P[g]     = (_Float16)p0;                   // A-layout: keys 8*hi+e
    P[8 + g] = (_Float16)p1;                   // keys 16+8*hi+e
  }
  ps += __shfl_xor(ps, 16, 32);
  l = l * alpha + ps;
  m = mn;
}

// broadcast per-row value (row = element g+8*hi) into a v8f via bpermute
__device__ __forceinline__ v8f row_bcast(float v, int hi) {
  v8f r;
  #pragma unroll
  for (int g = 0; g < 8; ++g) r[g] = __shfl(v, hi * 8 + g, 32);
  return r;
}

__global__ void __launch_bounds__(64, 1)
attn_kernel(const _Float16* __restrict__ q, const _Float16* __restrict__ k,
            const _Float16* __restrict__ vT, float* __restrict__ out) {
  const int lane = threadIdx.x & 31;
  const int wave = threadIdx.x >> 5;
  const int tile = blockIdx.x * 2 + wave;      // 0..511
  const int b    = tile / (Tc / 32);
  const int tq0  = (tile % (Tc / 32)) * 32;    // 32-aligned
  const int lrow = lane & 15;
  const int hi   = lane >> 4;
  const int tqa  = tq0 + lrow;                 // tile-a query of this lane
  const int tqb  = tq0 + 16 + lrow;            // tile-b query of this lane

  // Q as B-operand of S^T = K*Q^T: lane = query col, elements = head dims.
  const _Float16* qra = q + ((size_t)b * Tc + tqa) * Hc;
  const _Float16* qrb = q + ((size_t)b * Tc + tqb) * Hc;
  const v16h QB0a = load2x8h(qra + hi * 8, qra + hi * 8 + 16);
  const v16h QB1a = load2x8h(qra + 32 + hi * 8, qra + 32 + hi * 8 + 16);
  const v16h QB0b = load2x8h(qrb + hi * 8, qrb + hi * 8 + 16);
  const v16h QB1b = load2x8h(qrb + 32 + hi * 8, qrb + 32 + hi * 8 + 16);

  v8f ya0 = (v8f){}, ya1 = (v8f){}, ya2 = (v8f){}, ya3 = (v8f){};
  v8f yb0 = (v8f){}, yb1 = (v8f){}, yb2 = (v8f){}, yb3 = (v8f){};
  float ma = -1e30f, la = 0.0f, mb = -1e30f, lb = 0.0f;

  const int nblk = tq0 / 32 + 1;               // keys 0 .. tq0+31
  const _Float16* kb = k  + (size_t)b * Tc * Hc;
  const _Float16* vb = vT + (size_t)b * Hc * Tc;

  KVRegs cur = load_kv(kb, vb, 0, lrow, hi);

  #pragma unroll 2
  for (int blk = 0; blk < nblk; ++blk) {
    const int s0 = blk * 32;
    // prefetch next block's K/V (clamped, branchless), pinned here
    const int nb = (blk + 1 < nblk) ? (blk + 1) : blk;
    KVRegs nxt = load_kv(kb, vb, nb * 32, lrow, hi);
    __builtin_amdgcn_sched_barrier(0);

    // S^T = K * Q^T : lane = query, element g = key s0 + g + 8*hi (+16 for S1)
    v8f S0a = (v8f){}, S1a = (v8f){}, S0b = (v8f){}, S1b = (v8f){};
    S0a = wmma_f16(cur.ka00, QB0a, S0a); S0a = wmma_f16(cur.ka01, QB1a, S0a);
    S1a = wmma_f16(cur.ka10, QB0a, S1a); S1a = wmma_f16(cur.ka11, QB1a, S1a);
    S0b = wmma_f16(cur.ka00, QB0b, S0b); S0b = wmma_f16(cur.ka01, QB1b, S0b);
    S1b = wmma_f16(cur.ka10, QB0b, S1b); S1b = wmma_f16(cur.ka11, QB1b, S1b);

    if (blk == nblk - 1) {                     // only last block is masked
      #pragma unroll
      for (int g = 0; g < 8; ++g) {
        if (s0 + g + 8 * hi > tqa)      S0a[g] = -1e30f;
        if (s0 + 16 + g + 8 * hi > tqa) S1a[g] = -1e30f;
        if (s0 + g + 8 * hi > tqb)      S0b[g] = -1e30f;
        if (s0 + 16 + g + 8 * hi > tqb) S1b[g] = -1e30f;
      }
    }

    v16h Pa, Pb;
    float alpha_a, alpha_b;
    online_softmax(S0a, S1a, ma, la, Pa, alpha_a);
    online_softmax(S0b, S1b, mb, lb, Pb, alpha_b);
    const v8f ava = row_bcast(alpha_a, hi);
    const v8f avb = row_bcast(alpha_b, hi);
    ya0 *= ava; ya1 *= ava; ya2 *= ava; ya3 *= ava;
    yb0 *= avb; yb1 *= avb; yb2 *= avb; yb3 *= avb;

    ya0 = wmma_f16(Pa, cur.vb0, ya0); ya1 = wmma_f16(Pa, cur.vb1, ya1);
    ya2 = wmma_f16(Pa, cur.vb2, ya2); ya3 = wmma_f16(Pa, cur.vb3, ya3);
    yb0 = wmma_f16(Pb, cur.vb0, yb0); yb1 = wmma_f16(Pb, cur.vb1, yb1);
    yb2 = wmma_f16(Pb, cur.vb2, yb2); yb3 = wmma_f16(Pb, cur.vb3, yb3);

    cur = nxt;
  }

  // normalize by l (per query row, row = element g+8*hi) and store
  const v8f iva = row_bcast(__builtin_amdgcn_rcpf(la), hi);
  const v8f ivb = row_bcast(__builtin_amdgcn_rcpf(lb), hi);
  ya0 *= iva; ya1 *= iva; ya2 *= iva; ya3 *= iva;
  yb0 *= ivb; yb1 *= ivb; yb2 *= ivb; yb3 *= ivb;
  #pragma unroll
  for (int g = 0; g < 8; ++g) {
    const size_t oa = ((size_t)b * Tc + tq0 + g + 8 * hi) * Hc;
    const size_t ob = oa + (size_t)16 * Hc;
    out[oa +  0 + lrow] = ya0[g];
    out[oa + 16 + lrow] = ya1[g];
    out[oa + 32 + lrow] = ya2[g];
    out[oa + 48 + lrow] = ya3[g];
    out[ob +  0 + lrow] = yb0[g];
    out[ob + 16 + lrow] = yb1[g];
    out[ob + 32 + lrow] = yb2[g];
    out[ob + 48 + lrow] = yb3[g];
  }
}

// ---------------------------------------------------------------------------
extern "C" void kernel_launch(void* const* d_in, const int* in_sizes, int n_in,
                              void* d_out, int out_size, void* d_ws, size_t ws_size,
                              hipStream_t stream) {
  const float* x  = (const float*)d_in[0];
  const float* Wq = (const float*)d_in[1];
  const float* Wk = (const float*)d_in[2];
  const float* Wv = (const float*)d_in[3];
  float* out = (float*)d_out;

  char* ws = (char*)d_ws;
  const size_t WT_BYTES = (size_t)3 * Hc * Cc * sizeof(_Float16);   // 384 KB
  const size_t QK_BYTES = (size_t)Bc * Tc * Hc * sizeof(_Float16);  // 2 MB each
  _Float16* WT = (_Float16*)ws;
  _Float16* q  = (_Float16*)(ws + WT_BYTES);
  _Float16* kk = (_Float16*)(ws + WT_BYTES + QK_BYTES);
  _Float16* vT = (_Float16*)(ws + WT_BYTES + 2 * QK_BYTES);

  wt_kernel<<<(3 * Hc * Cc) / 256, 256, 0, stream>>>(Wq, Wk, Wv, WT);
  proj_kernel<<<(Bc * Tc / 16) / 4, 128, 0, stream>>>(x, WT, q, kk, vT);
  attn_kernel<<<(Bc * Tc / 32) / 2, 64, 0, stream>>>(q, kk, vT, out);
}
